// EdgeOperation_17815524344038
// MI455X (gfx1250) — matrix operand order repaired and checked by reading, not compiled, not run on hardware
//
#include <hip/hip_runtime.h>
#include <hip/hip_bf16.h>
#include <math.h>

typedef __attribute__((ext_vector_type(16))) _Float16 v16h;
typedef __attribute__((ext_vector_type(8)))  _Float16 v8h;
typedef __attribute__((ext_vector_type(4)))  _Float16 v4h;
typedef __attribute__((ext_vector_type(8)))  float    v8f;

// ---------------------------------------------------------------------------
// Padded concatenated input row (shared by GEMM1 and GEMM3):
//   [0..127]    f_i
//   [128..255]  f_j
//   [256..383]  m_ji          (hole for GEMM1; K-tiles 8..11 skipped there)
//   [384..511]  edge_feats
//   [512]       abs_r
//   [513..528]  rbf[0..15]
//   [529..543]  zero pad
// 17 K-tiles of 32. GEMM1 uses 13 tiles, GEMM3 uses all 17.
//
// Tiling: 64 edges/workgroup (4 M-tiles), 8 waves; wave w owns N-tile w and
// computes ALL 4 M-tiles -> each B-fragment load feeds 4 WMMAs (4x L2 reuse).
// ---------------------------------------------------------------------------

#define TILE_E 64
#define SA 552   // A_buf row stride in halves (276 DW = 20 mod 64 -> conflict-free b128)
#define SH 136   // H_buf row stride in halves (68 DW = 4 mod 64 -> conflict-free b128)

// Weight swizzle: B-fragment order. dst[((kt*8+nt)*32+lane)*16 + j] holds
// f16(W[row][n]) with n = nt*16 + (lane&15), K-pos p = kt*32 + (lane<16?0:8) + (j&7) + ((j&8)<<1).
__global__ void swizzle_w_kernel(const float* __restrict__ W,
                                 _Float16* __restrict__ dst,
                                 int total, int mode) {
    int idx = blockIdx.x * 256 + threadIdx.x;
    if (idx >= total) return;
    int j    = idx & 15;
    int lane = (idx >> 4) & 31;
    int nt   = (idx >> 9) & 7;
    int kt   = idx >> 12;
    int n = nt * 16 + (lane & 15);
    int p = kt * 32 + ((lane < 16) ? 0 : 8) + (j & 7) + ((j & 8) << 1);
    int row = -1;
    if (mode == 0) {            // W1_e: h_e = [f_i, f_j, abs_r, rbf, edge] (401 rows)
        if (p < 256)                 row = p;        // f_i, f_j
        else if (p >= 384 && p < 512) row = p - 111; // edge -> rows 273..400
        else if (p == 512)           row = 256;      // abs_r
        else if (p >= 513 && p < 529) row = p - 256; // rbf -> rows 257..272
    } else if (mode == 1) {     // W1_b: h_b = [f_i, f_j, m_ji, abs_r, rbf, edge] (529 rows)
        if (p < 384)                 row = p;        // f_i, f_j, m_ji
        else if (p < 512)            row = p + 17;   // edge -> rows 401..528
        else if (p == 512)           row = 384;      // abs_r
        else if (p < 529)            row = p - 128;  // rbf -> rows 385..400
    } else {                    // W2_e / W2_b: 128 rows
        if (p < 128)                 row = p;
    }
    float v = (row >= 0) ? W[(size_t)row * 128 + n] : 0.0f;
    dst[idx] = (_Float16)v;
}

// A-fragment (16x32 f16): halves 0..7 at p, halves 8..15 at p+16 (two b128 LDS reads)
__device__ __forceinline__ v16h ldA(const _Float16* p) {
    v8h lo = *(const v8h*)p;
    v8h hi = *(const v8h*)(p + 16);
    v16h r;
#pragma unroll
    for (int i = 0; i < 8; ++i) { r[i] = lo[i]; r[i + 8] = hi[i]; }
    return r;
}

__device__ __forceinline__ v8f wmma16(v16h a, v16h b, v8f c) {
    return __builtin_amdgcn_wmma_f32_16x16x32_f16(false, a, false, b, (short)0, c,
                                                  false, false);
}

__device__ __forceinline__ float silu(float x) {
    return x / (1.0f + __expf(-x));
}

__global__ __launch_bounds__(256)
void edge_mlp_kernel(const float* __restrict__ nf,      // node_feats [N,128]
                     const float* __restrict__ coords,  // [N,3]
                     const float* __restrict__ ef,      // edge_feats [E,128]
                     const int*   __restrict__ eidx,    // [2,E]
                     const float* __restrict__ cutoff,  // [1]
                     const float* __restrict__ z0k,     // [16]
                     const _Float16* __restrict__ W1e, const float* __restrict__ b1e,
                     const _Float16* __restrict__ W2e, const float* __restrict__ b2e,
                     const _Float16* __restrict__ W1b, const float* __restrict__ b1b,
                     const _Float16* __restrict__ W2b, const float* __restrict__ b2b,
                     float* __restrict__ out, int E) {
    __shared__ _Float16 A_buf[TILE_E * SA];   // 70656 B
    __shared__ _Float16 H_buf[TILE_E * SH];   // 17408 B

    const int  t  = threadIdx.x;
    const long e0 = (long)blockIdx.x * TILE_E;

    // ---------------- Phase 0: build concatenated row in LDS (f16) ----------
    {
        int el = t >> 2;            // edge within tile, 4 threads/edge
        int l4 = t & 3;
        long e = e0 + el;
        if (e >= E) e = E - 1;      // clamp (stores are guarded later)
        int s  = eidx[e];
        int rv = eidx[(long)E + e];
        const float* fi = nf + (size_t)s * 128;
        const float* fj = nf + (size_t)rv * 128;
        const float* fe = ef + (size_t)e * 128;
        _Float16* row = A_buf + el * SA;
#pragma unroll
        for (int q = 0; q < 8; ++q) {
            int c = l4 * 32 + q * 4;
            float4 a = *(const float4*)(fi + c);
            float4 b = *(const float4*)(fj + c);
            float4 g = *(const float4*)(fe + c);
            v4h ha; ha[0] = (_Float16)a.x; ha[1] = (_Float16)a.y;
                    ha[2] = (_Float16)a.z; ha[3] = (_Float16)a.w;
            v4h hb; hb[0] = (_Float16)b.x; hb[1] = (_Float16)b.y;
                    hb[2] = (_Float16)b.z; hb[3] = (_Float16)b.w;
            v4h hg; hg[0] = (_Float16)g.x; hg[1] = (_Float16)g.y;
                    hg[2] = (_Float16)g.z; hg[3] = (_Float16)g.w;
            *(v4h*)(row + c)       = ha;
            *(v4h*)(row + 128 + c) = hb;
            *(v4h*)(row + 384 + c) = hg;
        }
        if (l4 == 0) {
            float dx = coords[3 * (size_t)s + 0] - coords[3 * (size_t)rv + 0];
            float dy = coords[3 * (size_t)s + 1] - coords[3 * (size_t)rv + 1];
            float dz = coords[3 * (size_t)s + 2] - coords[3 * (size_t)rv + 2];
            float r  = sqrtf(dx * dx + dy * dy + dz * dz);
            float c  = cutoff[0];
            float pref = sqrtf(2.0f / c);
            float inv  = 1.0f / r;
            row[512] = (_Float16)r;
#pragma unroll
            for (int k = 0; k < 16; ++k)
                row[513 + k] = (_Float16)(pref * __sinf(z0k[k] * r / c) * inv);
        }
        if (l4 == 1) {
#pragma unroll
            for (int k = 529; k < 544; ++k) row[k] = (_Float16)0.0f;
        }
    }
    __syncthreads();

    // ---------------- WMMA tiling ------------------------------------------
    const int w    = t >> 5;           // wave 0..7
    const int L    = t & 31;           // lane
    const int nt   = w;                // N-tile owned by this wave
    const int lrow = L & 15;
    const int klo  = (L < 16) ? 0 : 8;
    const int rb   = (L < 16) ? 0 : 8;
    const int col  = nt * 16 + lrow;

    // ---------------- GEMM1: silu(h_e @ W1_e + b1_e) -> H_buf --------------
    {
        v8f acc[4] = {};
#pragma unroll
        for (int kt = 0; kt < 17; ++kt) {
            if (kt >= 8 && kt <= 11) continue;           // m_ji hole
            const _Float16* ap = A_buf + lrow * SA + kt * 32 + klo;
            v16h b = *(const v16h*)(W1e + ((((kt * 8 + nt) * 32) + L) << 4));
#pragma unroll
            for (int mm = 0; mm < 4; ++mm)
                acc[mm] = wmma16(ldA(ap + mm * 16 * SA), b, acc[mm]);
        }
        float bb = b1e[col];
#pragma unroll
        for (int mm = 0; mm < 4; ++mm)
#pragma unroll
            for (int r = 0; r < 8; ++r)
                H_buf[(mm * 16 + rb + r) * SH + col] =
                    (_Float16)silu(acc[mm][r] + bb);
    }
    __syncthreads();

    // ---------------- GEMM2: hidden @ W2_e + b2_e = m_ji -------------------
    {
        v8f acc[4] = {};
#pragma unroll
        for (int kt = 0; kt < 4; ++kt) {
            const _Float16* ap = H_buf + lrow * SH + kt * 32 + klo;
            v16h b = *(const v16h*)(W2e + ((((kt * 8 + nt) * 32) + L) << 4));
#pragma unroll
            for (int mm = 0; mm < 4; ++mm)
                acc[mm] = wmma16(ldA(ap + mm * 16 * SH), b, acc[mm]);
        }
        float bb = b2e[col];
#pragma unroll
        for (int mm = 0; mm < 4; ++mm) {
#pragma unroll
            for (int r = 0; r < 8; ++r) {
                int mr = mm * 16 + rb + r;
                float y = acc[mm][r] + bb;
                A_buf[mr * SA + 256 + col] = (_Float16)y;   // fill h_b hole
                long e = e0 + mr;
                if (e < E) out[(size_t)e * 128 + col] = y;  // m_ji output
            }
        }
    }
    __syncthreads();

    // ---------------- GEMM3: silu(h_b @ W1_b + b1_b) -> H_buf --------------
    {
        v8f acc[4] = {};
#pragma unroll
        for (int kt = 0; kt < 17; ++kt) {
            const _Float16* ap = A_buf + lrow * SA + kt * 32 + klo;
            v16h b = *(const v16h*)(W1b + ((((kt * 8 + nt) * 32) + L) << 4));
#pragma unroll
            for (int mm = 0; mm < 4; ++mm)
                acc[mm] = wmma16(ldA(ap + mm * 16 * SA), b, acc[mm]);
        }
        float bb = b1b[col];
#pragma unroll
        for (int mm = 0; mm < 4; ++mm)
#pragma unroll
            for (int r = 0; r < 8; ++r)
                H_buf[(mm * 16 + rb + r) * SH + col] =
                    (_Float16)silu(acc[mm][r] + bb);
    }
    __syncthreads();

    // ---------------- GEMM4: hidden @ W2_b + b2_b = edge_out ---------------
    {
        v8f acc[4] = {};
#pragma unroll
        for (int kt = 0; kt < 4; ++kt) {
            const _Float16* ap = H_buf + lrow * SH + kt * 32 + klo;
            v16h b = *(const v16h*)(W2b + ((((kt * 8 + nt) * 32) + L) << 4));
#pragma unroll
            for (int mm = 0; mm < 4; ++mm)
                acc[mm] = wmma16(ldA(ap + mm * 16 * SH), b, acc[mm]);
        }
        float bb = b2b[col];
        float* out2 = out + (size_t)E * 128;
#pragma unroll
        for (int mm = 0; mm < 4; ++mm) {
#pragma unroll
            for (int r = 0; r < 8; ++r) {
                int mr = mm * 16 + rb + r;
                long e = e0 + mr;
                if (e < E) out2[(size_t)e * 128 + col] = acc[mm][r] + bb;
            }
        }
    }
}

extern "C" void kernel_launch(void* const* d_in, const int* in_sizes, int n_in,
                              void* d_out, int out_size, void* d_ws, size_t ws_size,
                              hipStream_t stream) {
    const float* nf     = (const float*)d_in[0];
    const float* coords = (const float*)d_in[1];
    const float* ef     = (const float*)d_in[2];
    const int*   eidx   = (const int*)d_in[3];
    const float* cutoff = (const float*)d_in[4];
    const float* z0k    = (const float*)d_in[5];
    const float* W1e_f  = (const float*)d_in[6];
    const float* b1e    = (const float*)d_in[7];
    const float* W2e_f  = (const float*)d_in[8];
    const float* b2e    = (const float*)d_in[9];
    const float* W1b_f  = (const float*)d_in[10];
    const float* b1b    = (const float*)d_in[11];
    const float* W2b_f  = (const float*)d_in[12];
    const float* b2b    = (const float*)d_in[13];

    const int E = in_sizes[2] / 128;

    // Workspace: f16 swizzled weights (17 and 4 K-tiles x 8 N-tiles x 512 halves)
    const size_t SZ17 = (size_t)17 * 8 * 32 * 16 * sizeof(_Float16); // 139264 B
    const size_t SZ4  = (size_t)4  * 8 * 32 * 16 * sizeof(_Float16); //  32768 B
    char* ws = (char*)d_ws;
    _Float16* W1e = (_Float16*)(ws);
    _Float16* W2e = (_Float16*)(ws + SZ17);
    _Float16* W1b = (_Float16*)(ws + SZ17 + SZ4);
    _Float16* W2b = (_Float16*)(ws + 2 * SZ17 + SZ4);

    const int T17 = 17 * 8 * 32 * 16;   // 69632
    const int T4  = 4  * 8 * 32 * 16;   // 16384
    swizzle_w_kernel<<<(T17 + 255) / 256, 256, 0, stream>>>(W1e_f, W1e, T17, 0);
    swizzle_w_kernel<<<(T4  + 255) / 256, 256, 0, stream>>>(W2e_f, W2e, T4, 2);
    swizzle_w_kernel<<<(T17 + 255) / 256, 256, 0, stream>>>(W1b_f, W1b, T17, 1);
    swizzle_w_kernel<<<(T4  + 255) / 256, 256, 0, stream>>>(W2b_f, W2b, T4, 2);

    const int nblocks = (E + TILE_E - 1) / TILE_E;
    edge_mlp_kernel<<<nblocks, 256, 0, stream>>>(nf, coords, ef, eidx, cutoff, z0k,
                                                 W1e, b1e, W2e, b2e,
                                                 W1b, b1b, W2b, b2b,
                                                 (float*)d_out, E);
}
